// Block_2637109920380
// MI455X (gfx1250) — matrix operand order repaired and checked by reading, not compiled, not run on hardware
//
#include <hip/hip_runtime.h>
#include <math.h>

#define B_  4
#define T_  2048
#define E_  1024
#define H_  16
#define DH_ 64

typedef __attribute__((ext_vector_type(16))) __bf16 v16bf;
typedef __attribute__((ext_vector_type(8)))  __bf16 v8bf;
typedef __attribute__((ext_vector_type(8)))  float  v8f;
typedef int v4i __attribute__((vector_size(16)));   // matches builtin param pointee

__device__ __forceinline__ __bf16 to_bf(float f) {
  unsigned int u = __builtin_bit_cast(unsigned int, f);
  unsigned int r = (u + 0x7FFFu + ((u >> 16) & 1u)) >> 16;
  unsigned short s = (unsigned short)r;
  return __builtin_bit_cast(__bf16, s);
}

// A-fragment (16-bit A 16x32, wave32): lane holds row m=lane&15,
// K elements {hi*16 .. hi*16+15}  (hi = lane>>4)
__device__ __forceinline__ v16bf load_a_frag(const __bf16* row, int hi) {
  v8bf lo = *(const v8bf*)(row + hi * 16);
  v8bf hv = *(const v8bf*)(row + hi * 16 + 8);
  v16bf r;
#pragma unroll
  for (int i = 0; i < 8; ++i) { r[i] = lo[i]; r[i + 8] = hv[i]; }
  return r;
}

// B-fragment: lane holds 16 contiguous K values of one column (16B-aligned source)
__device__ __forceinline__ v16bf load_b_frag(const __bf16* p) {
  v8bf lo = *(const v8bf*)p;
  v8bf hv = *(const v8bf*)(p + 8);
  v16bf r;
#pragma unroll
  for (int i = 0; i < 8; ++i) { r[i] = lo[i]; r[i + 8] = hv[i]; }
  return r;
}

// ---- async global->LDS copy (16B), ASYNCcnt-tracked; guarded fallback ----
#if __has_builtin(__builtin_amdgcn_global_load_async_to_lds_b128)
#define ASYNC_CP16(g, l)                                                      \
  __builtin_amdgcn_global_load_async_to_lds_b128(                             \
      (__attribute__((address_space(1))) v4i*)(g),                            \
      (__attribute__((address_space(3))) v4i*)(l), 0, 0)
#if __has_builtin(__builtin_amdgcn_s_wait_asynccnt)
#define ASYNC_WAIT() __builtin_amdgcn_s_wait_asynccnt(0)
#else
#define ASYNC_WAIT() asm volatile("s_wait_asynccnt 0" ::: "memory")
#endif
#else
#define ASYNC_CP16(g, l) (*(v8bf*)(l) = *(const v8bf*)(g))
#define ASYNC_WAIT() ((void)0)
#endif

// ---------------- LayerNorm (fp32 in -> bf16 out) ----------------
__global__ void ln_bf16_kernel(const float* __restrict__ x, const float* __restrict__ g,
                               const float* __restrict__ b, __bf16* __restrict__ out) {
  const int row = blockIdx.x;
  const int tid = threadIdx.x;                       // 256 threads, 4 elems each
  const float4 v = ((const float4*)(x + (size_t)row * E_))[tid];
  float s  = v.x + v.y + v.z + v.w;
  float ss = v.x * v.x + v.y * v.y + v.z * v.z + v.w * v.w;
#pragma unroll
  for (int o = 16; o >= 1; o >>= 1) { s += __shfl_xor(s, o, 32); ss += __shfl_xor(ss, o, 32); }
  __shared__ float shs[8], shss[8];
  const int wave = tid >> 5, lane = tid & 31;
  if (lane == 0) { shs[wave] = s; shss[wave] = ss; }
  __syncthreads();
  float ts = 0.f, tss = 0.f;
#pragma unroll
  for (int i = 0; i < 8; ++i) { ts += shs[i]; tss += shss[i]; }
  const float mean = ts * (1.0f / E_);
  const float var  = tss * (1.0f / E_) - mean * mean;
  const float rstd = rsqrtf(var + 1e-5f);
  const int c0 = tid * 4;
  __bf16* o = out + (size_t)row * E_ + c0;
  o[0] = to_bf((v.x - mean) * rstd * g[c0 + 0] + b[c0 + 0]);
  o[1] = to_bf((v.y - mean) * rstd * g[c0 + 1] + b[c0 + 1]);
  o[2] = to_bf((v.z - mean) * rstd * g[c0 + 2] + b[c0 + 2]);
  o[3] = to_bf((v.w - mean) * rstd * g[c0 + 3] + b[c0 + 3]);
}

// ---------- Wq/Wk/Wv [H,E,DH] fp32 -> WT [E_out=H*DH, E] bf16 (transposed) ----------
__global__ void repack_qkv_kernel(const float* __restrict__ W, __bf16* __restrict__ WT) {
  const int idx = blockIdx.x * 256 + threadIdx.x;    // over E_*E_ = 1M
  const int n = idx >> 10;                           // output col = h*64+d
  const int e = idx & 1023;
  const int h = n >> 6, d = n & 63;
  WT[idx] = to_bf(W[((size_t)h * E_ + e) * DH_ + d]);
}

// ---------- generic W [K,N] fp32 -> WT [N,K] bf16 ----------
__global__ void transpose_w_kernel(const float* __restrict__ W, __bf16* __restrict__ WT,
                                   int K, int N) {
  const int idx = blockIdx.x * 256 + threadIdx.x;    // over N*K
  const int n = idx / K;
  const int k = idx - n * K;
  WT[(size_t)n * K + k] = to_bf(W[(size_t)k * N + n]);
}

// ---------- v [B*T, E] bf16 -> Vt [B,H,DH,T] bf16 ----------
__global__ void transpose_v_kernel(const __bf16* __restrict__ v, __bf16* __restrict__ vt) {
  const int idx = blockIdx.x * 256 + threadIdx.x;    // over B*T*E = 8M
  const int col = idx & 1023;                        // h*64+d
  const int row = idx >> 10;                         // b*T+t
  const int b = row >> 11, t = row & 2047;
  const int h = col >> 6, d = col & 63;
  vt[(((size_t)(b * H_ + h) * DH_ + d) << 11) + t] = v[idx];
}

// ------- bf16 WMMA GEMM: C[M,N] = A[M,K] x BT[N,K]^T (+bias,+resid,ReLU) -------
// Block tile 128x64 (8 waves as 4M x 2N), wave tile 32x32 (4 accumulators).
// A/B tiles double-buffered in LDS via async global->LDS copies.
#define LDSP 40   // padded LDS row stride (32 data + 8 pad bf16) => 80B, 16B aligned
__global__ __launch_bounds__(256) void gemm_bf16_kernel(
    const __bf16* __restrict__ A, const __bf16* __restrict__ BT,
    float* __restrict__ outf, __bf16* __restrict__ outb,
    const float* __restrict__ bias, const float* __restrict__ resid,
    int M, int N, int K, int do_relu) {
  __shared__ __align__(32) __bf16 Ab[2][128][LDSP];
  __shared__ __align__(32) __bf16 Bb[2][64][LDSP];
  const int tid = threadIdx.x;
  const int lane = tid & 31, wave = tid >> 5;
  const int wm = wave & 3, wn = wave >> 2;           // 4 M-waves x 2 N-waves
  const int l15 = lane & 15, hi = lane >> 4;
  const int mblk = blockIdx.y * 128;
  const int nblk = blockIdx.x * 64;

  // per-thread 16B copy chunks: A tile = 512 chunks (2/thread), B tile = 256 (1/thread)
  const int crow = tid >> 2, cseg = (tid & 3) * 8;
  const __bf16* Ag0 = A + (size_t)(mblk + crow) * K + cseg;
  const __bf16* Ag1 = A + (size_t)(mblk + 64 + crow) * K + cseg;
  const __bf16* Bg  = BT + (size_t)(nblk + crow) * K + cseg;

  auto issue = [&](int buf, int k0) {
    __builtin_prefetch(Ag0 + k0 + 64, 0, 0);         // warm GL2 two steps ahead
    ASYNC_CP16(Ag0 + k0, &Ab[buf][crow][cseg]);
    ASYNC_CP16(Ag1 + k0, &Ab[buf][64 + crow][cseg]);
    ASYNC_CP16(Bg + k0,  &Bb[buf][crow][cseg]);
  };

  v8f acc[2][2] = {{{}, {}}, {{}, {}}};
  const int nk = K >> 5;
  issue(0, 0);
  for (int ks = 0; ks < nk; ++ks) {
    const int cur = ks & 1;
    ASYNC_WAIT();                                    // my copies for buf 'cur' done
    __syncthreads();                                 // all waves' copies done + prev reads done
    if (ks + 1 < nk) issue(cur ^ 1, (ks + 1) << 5);  // overlap next copy with compute
    const v16bf a0 = load_a_frag(&Ab[cur][wm * 32 + l15][0], hi);
    const v16bf a1 = load_a_frag(&Ab[cur][wm * 32 + 16 + l15][0], hi);
    const v16bf b0 = load_b_frag(&Bb[cur][wn * 32 + l15][hi * 16]);
    const v16bf b1 = load_b_frag(&Bb[cur][wn * 32 + 16 + l15][hi * 16]);
    acc[0][0] = __builtin_amdgcn_wmma_f32_16x16x32_bf16(false, a0, false, b0, (short)0, acc[0][0], false, false);
    acc[0][1] = __builtin_amdgcn_wmma_f32_16x16x32_bf16(false, a0, false, b1, (short)0, acc[0][1], false, false);
    acc[1][0] = __builtin_amdgcn_wmma_f32_16x16x32_bf16(false, a1, false, b0, (short)0, acc[1][0], false, false);
    acc[1][1] = __builtin_amdgcn_wmma_f32_16x16x32_bf16(false, a1, false, b1, (short)0, acc[1][1], false, false);
  }
#pragma unroll
  for (int am = 0; am < 2; ++am)
#pragma unroll
    for (int bn = 0; bn < 2; ++bn) {
      const int n = nblk + wn * 32 + bn * 16 + l15;
      const float bv = bias ? bias[n] : 0.0f;
#pragma unroll
      for (int r = 0; r < 8; ++r) {
        const size_t row = (size_t)(mblk + wm * 32 + am * 16 + r + hi * 8);
        float v = acc[am][bn][r] + bv;
        if (resid) v += resid[row * N + n];
        if (do_relu) v = fmaxf(v, 0.0f);
        if (outf) outf[row * N + n] = v;
        else      outb[row * N + n] = to_bf(v);
      }
    }
}

// ---------------- causal flash attention (one wave per 16 query rows) ----------------
__global__ void attn_kernel(const __bf16* __restrict__ q, const __bf16* __restrict__ k,
                            const __bf16* __restrict__ vt, __bf16* __restrict__ att) {
  __shared__ __bf16 lds_p[8][16][32];                // per-wave P tile
  const int lane = threadIdx.x & 31, wave = threadIdx.x >> 5;
  const int l15 = lane & 15, hi = lane >> 4;
  const int bh = blockIdx.y;                         // b*H + h
  const int b = bh >> 4, h = bh & 15;
  const int t0 = (blockIdx.x * 8 + wave) * 16;
  const float scale = 0.03125f;                      // 1/sqrt(E)

  const __bf16* qrow = q + (size_t)(b * T_ + t0 + l15) * E_ + h * DH_;
  const v16bf qa0 = load_a_frag(qrow, hi);           // d 0..31
  const v16bf qa1 = load_a_frag(qrow + 32, hi);      // d 32..63

  float mrow[8], lrow[8];
  v8f acc[4] = {{}, {}, {}, {}};
#pragma unroll
  for (int r = 0; r < 8; ++r) { mrow[r] = -__builtin_inff(); lrow[r] = 0.0f; }

  for (int s0 = 0; s0 <= t0 + 15; s0 += 32) {        // wave-uniform causal loop
    v8f sf[2];
#pragma unroll
    for (int cix = 0; cix < 2; ++cix) {
      int s = s0 + cix * 16 + l15;
      if (s > T_ - 1) s = T_ - 1;                    // clamp (masked anyway)
      const __bf16* kp = k + (size_t)(b * T_ + s) * E_ + h * DH_ + hi * 16;
      v16bf kb0 = load_b_frag(kp);
      v16bf kb1 = load_b_frag(kp + 32);
      v8f z = {};
      z = __builtin_amdgcn_wmma_f32_16x16x32_bf16(false, qa0, false, kb0, (short)0, z, false, false);
      z = __builtin_amdgcn_wmma_f32_16x16x32_bf16(false, qa1, false, kb1, (short)0, z, false, false);
      sf[cix] = z;
    }
    float sv0[8], sv1[8], rm[8];
#pragma unroll
    for (int r = 0; r < 8; ++r) {
      const int t = t0 + r + hi * 8;
      sv0[r] = (s0 + l15      <= t) ? sf[0][r] * scale : -__builtin_inff();
      sv1[r] = (s0 + 16 + l15 <= t) ? sf[1][r] * scale : -__builtin_inff();
      rm[r] = fmaxf(sv0[r], sv1[r]);
    }
#pragma unroll
    for (int r = 0; r < 8; ++r)
#pragma unroll
      for (int o = 1; o < 16; o <<= 1) rm[r] = fmaxf(rm[r], __shfl_xor(rm[r], o, 32));
    float al[8], rs[8];
#pragma unroll
    for (int r = 0; r < 8; ++r) {
      const float mnew = fmaxf(mrow[r], rm[r]);
      al[r] = __expf(mrow[r] - mnew);
      const float p0 = __expf(sv0[r] - mnew);
      const float p1 = __expf(sv1[r] - mnew);
      lds_p[wave][r + hi * 8][l15]      = to_bf(p0);
      lds_p[wave][r + hi * 8][16 + l15] = to_bf(p1);
      rs[r] = p0 + p1;
      mrow[r] = mnew;
    }
#pragma unroll
    for (int r = 0; r < 8; ++r)
#pragma unroll
      for (int o = 1; o < 16; o <<= 1) rs[r] += __shfl_xor(rs[r], o, 32);
#pragma unroll
    for (int r = 0; r < 8; ++r) {
      lrow[r] = lrow[r] * al[r] + rs[r];
#pragma unroll
      for (int dc = 0; dc < 4; ++dc) acc[dc][r] *= al[r];
    }
    asm volatile("s_wait_dscnt 0" ::: "memory");     // LDS store -> load, same wave
    const v16bf pa = load_a_frag(&lds_p[wave][l15][0], hi);
    int sb = s0 + hi * 16;
    if (sb > T_ - 16) sb = T_ - 16;                  // clamp; P cols there are 0
#pragma unroll
    for (int dc = 0; dc < 4; ++dc) {
      const __bf16* vp = vt + (size_t)(bh * DH_ + dc * 16 + l15) * T_ + sb;
      v16bf vb = load_b_frag(vp);
      acc[dc] = __builtin_amdgcn_wmma_f32_16x16x32_bf16(false, pa, false, vb, (short)0, acc[dc], false, false);
    }
  }
#pragma unroll
  for (int dc = 0; dc < 4; ++dc)
#pragma unroll
    for (int r = 0; r < 8; ++r) {
      const float o = acc[dc][r] / lrow[r];
      att[(size_t)(b * T_ + t0 + r + hi * 8) * E_ + h * DH_ + dc * 16 + l15] = to_bf(o);
    }
}

// ------------------------------- launcher -------------------------------
extern "C" void kernel_launch(void* const* d_in, const int* in_sizes, int n_in,
                              void* d_out, int out_size, void* d_ws, size_t ws_size,
                              hipStream_t stream) {
  (void)in_sizes; (void)n_in; (void)out_size; (void)ws_size;
  const float* x    = (const float*)d_in[0];
  const float* ln1g = (const float*)d_in[1];
  const float* ln1b = (const float*)d_in[2];
  const float* Wq   = (const float*)d_in[3];
  const float* Wk   = (const float*)d_in[4];
  const float* Wv   = (const float*)d_in[5];
  const float* Wp   = (const float*)d_in[6];
  const float* bp   = (const float*)d_in[7];
  const float* ln2g = (const float*)d_in[8];
  const float* ln2b = (const float*)d_in[9];
  const float* W1   = (const float*)d_in[10];
  const float* b1   = (const float*)d_in[11];
  const float* W2   = (const float*)d_in[12];
  const float* b2   = (const float*)d_in[13];
  float* out = (float*)d_out;

  char* ws = (char*)d_ws;
  size_t off = 0;
  auto alloc = [&](size_t bytes) -> char* {
    char* p = ws + off;
    off += (bytes + 255) & ~(size_t)255;
    return p;
  };
  const size_t MT = (size_t)B_ * T_;                 // 8192 rows
  __bf16* h1  = (__bf16*)alloc(MT * E_ * 2);
  __bf16* WqT = (__bf16*)alloc((size_t)E_ * E_ * 2);
  __bf16* WkT = (__bf16*)alloc((size_t)E_ * E_ * 2);
  __bf16* WvT = (__bf16*)alloc((size_t)E_ * E_ * 2);
  __bf16* qb  = (__bf16*)alloc(MT * E_ * 2);
  __bf16* kb  = (__bf16*)alloc(MT * E_ * 2);
  __bf16* vb  = (__bf16*)alloc(MT * E_ * 2);
  __bf16* vtb = (__bf16*)alloc(MT * E_ * 2);
  __bf16* att = (__bf16*)alloc(MT * E_ * 2);
  __bf16* WpT = (__bf16*)alloc((size_t)E_ * E_ * 2);
  float*  x1  = (float*)alloc(MT * E_ * 4);
  __bf16* h2  = (__bf16*)alloc(MT * E_ * 2);
  __bf16* W1T = (__bf16*)alloc((size_t)E_ * 4 * E_ * 2);
  __bf16* W2T = (__bf16*)alloc((size_t)E_ * 4 * E_ * 2);
  __bf16* ff1 = (__bf16*)alloc(MT * 4 * E_ * 2);

  const dim3 blk(256);
  // LN1
  ln_bf16_kernel<<<dim3((unsigned)MT), blk, 0, stream>>>(x, ln1g, ln1b, h1);
  // weight repacks (transposed bf16)
  repack_qkv_kernel<<<dim3((E_ * E_) / 256), blk, 0, stream>>>(Wq, WqT);
  repack_qkv_kernel<<<dim3((E_ * E_) / 256), blk, 0, stream>>>(Wk, WkT);
  repack_qkv_kernel<<<dim3((E_ * E_) / 256), blk, 0, stream>>>(Wv, WvT);
  transpose_w_kernel<<<dim3((E_ * E_) / 256), blk, 0, stream>>>(Wp, WpT, E_, E_);
  transpose_w_kernel<<<dim3((E_ * 4 * E_) / 256), blk, 0, stream>>>(W1, W1T, E_, 4 * E_);
  transpose_w_kernel<<<dim3((E_ * 4 * E_) / 256), blk, 0, stream>>>(W2, W2T, 4 * E_, E_);
  // QKV projections (bf16 WMMA, async-LDS staged)
  gemm_bf16_kernel<<<dim3(E_ / 64, MT / 128), blk, 0, stream>>>(h1, WqT, nullptr, qb, nullptr, nullptr, (int)MT, E_, E_, 0);
  gemm_bf16_kernel<<<dim3(E_ / 64, MT / 128), blk, 0, stream>>>(h1, WkT, nullptr, kb, nullptr, nullptr, (int)MT, E_, E_, 0);
  gemm_bf16_kernel<<<dim3(E_ / 64, MT / 128), blk, 0, stream>>>(h1, WvT, nullptr, vb, nullptr, nullptr, (int)MT, E_, E_, 0);
  // V -> [B,H,DH,T] for contiguous B-fragments in attention
  transpose_v_kernel<<<dim3((unsigned)(MT * E_ / 256)), blk, 0, stream>>>(vb, vtb);
  // causal flash attention
  attn_kernel<<<dim3(T_ / 128, B_ * H_), blk, 0, stream>>>(qb, kb, vtb, att);
  // output projection + bias + residual  -> x1 (fp32)
  gemm_bf16_kernel<<<dim3(E_ / 64, MT / 128), blk, 0, stream>>>(att, WpT, x1, nullptr, bp, x, (int)MT, E_, E_, 0);
  // LN2
  ln_bf16_kernel<<<dim3((unsigned)MT), blk, 0, stream>>>(x1, ln2g, ln2b, h2);
  // FFN
  gemm_bf16_kernel<<<dim3(4 * E_ / 64, MT / 128), blk, 0, stream>>>(h2, W1T, nullptr, ff1, b1, nullptr, (int)MT, 4 * E_, E_, 1);
  gemm_bf16_kernel<<<dim3(E_ / 64, MT / 128), blk, 0, stream>>>(ff1, W2T, out, nullptr, b2, x1, (int)MT, E_, 4 * E_, 0);
}
